// VQ_VAE_63462436765699
// MI455X (gfx1250) — matrix operand order repaired and checked by reading, not compiled, not run on hardware
//
#include <hip/hip_runtime.h>

// VQ-VAE nearest-codebook quantization for MI455X (gfx1250, wave32, WMMA).
// score[n,k] = ||e_k||^2 - 2 <x_n, e_k>   (x^2 constant per row -> dropped)
// dots via bf16 hi/lo split: x_hi*e_hi + x_hi*e_lo + x_lo*e_hi (~fp32 accurate)
//
// Fast path: prep kernel splits codebook to bf16 hi/lo + e2 in d_ws ONCE;
// main kernel keeps A-fragments resident in VGPRs, double-buffers B blocks in
// LDS, and stages them with async global->LDS (ASYNCcnt) when available.

typedef __attribute__((ext_vector_type(16))) __bf16 bf16x16;
typedef __attribute__((ext_vector_type(8)))  float  f32x8;

#define NQ    65536   // B*H*W
#define DDIM  256
#define KCODE 1024
#define MTILE 64      // query rows per workgroup
#define NBLK  32      // codes per block iteration
#define NITER (KCODE / NBLK)

#if defined(__gfx1250__) && __has_builtin(__builtin_amdgcn_global_load_async_to_lds_b128)
#define USE_ASYNC_LDS 1
#else
#define USE_ASYNC_LDS 0
#endif

#if USE_ASYNC_LDS
// Signature (from hipcc diagnostic): first param is
//   __attribute__((__vector_size__(4*sizeof(int)))) int __device__ *   (global int4*)
typedef __attribute__((__vector_size__(4 * sizeof(int)))) int vint4;
typedef __attribute__((address_space(1))) vint4 gvint4;   // global
typedef __attribute__((address_space(3))) vint4 lvint4;   // LDS
#endif

__device__ __forceinline__ unsigned short f32_to_bf16_rn(float f) {
  unsigned int u = __float_as_uint(f);
  u += 0x7FFFu + ((u >> 16) & 1u);          // round-to-nearest-even
  return (unsigned short)(u >> 16);
}
__device__ __forceinline__ float bf16_bits_to_f32(unsigned short h) {
  return __uint_as_float(((unsigned int)h) << 16);
}

union Frag { uint4 q[2]; bf16x16 v; };

__device__ __forceinline__ void wait_async_lds() {
#if USE_ASYNC_LDS
#if __has_builtin(__builtin_amdgcn_s_wait_asynccnt)
  __builtin_amdgcn_s_wait_asynccnt(0);
#else
  asm volatile("s_wait_asynccnt 0x0" ::: "memory");
#endif
#endif
}

// ---------------------------------------------------------------------------
// Prep: split codebook into bf16 hi/lo and compute e2, once per launch.
// ---------------------------------------------------------------------------
__global__ __launch_bounds__(256) void vq_prep_embeds(
    const float* __restrict__ embeds, unsigned short* __restrict__ eh,
    unsigned short* __restrict__ el, float* __restrict__ e2)
{
  __shared__ float red[256];
  const int k = blockIdx.x, t = threadIdx.x;
  float f = embeds[(size_t)k * DDIM + t];
  unsigned short h = f32_to_bf16_rn(f);
  eh[(size_t)k * DDIM + t] = h;
  el[(size_t)k * DDIM + t] = f32_to_bf16_rn(f - bf16_bits_to_f32(h));
  red[t] = f * f;
  __syncthreads();
  #pragma unroll
  for (int s = 128; s > 0; s >>= 1) {
    if (t < s) red[t] += red[t + s];
    __syncthreads();
  }
  if (t == 0) e2[k] = red[0];
}

// ---------------------------------------------------------------------------
// Fast main kernel: A in VGPRs, double-buffered B staging (async if possible).
// ---------------------------------------------------------------------------
__global__ __launch_bounds__(256) void vq_argmin_fast(
    const float* __restrict__ pred, const float* __restrict__ embeds,
    const unsigned short* __restrict__ eh, const unsigned short* __restrict__ el,
    const float* __restrict__ e2g,
    float* __restrict__ out_embeds, float* __restrict__ out_classes)
{
  __shared__ __align__(16) unsigned short Xhi[MTILE * DDIM];      // 32 KB
  __shared__ __align__(16) unsigned short Xlo[MTILE * DDIM];      // 32 KB
  __shared__ __align__(16) unsigned short Ehi[2][NBLK * DDIM];    // 32 KB
  __shared__ __align__(16) unsigned short Elo[2][NBLK * DDIM];    // 32 KB
  __shared__ float red_s[MTILE * 2];
  __shared__ int   red_i[MTILE * 2];
  __shared__ int   cls_lds[MTILE];

  const int t       = threadIdx.x;
  const int wg      = blockIdx.x;
  const int lane    = t & 31;
  const int wave    = t >> 5;
  const int wr      = wave >> 1;        // rows wr*16 .. wr*16+15
  const int wc      = wave & 1;         // codes wc*16 .. wc*16+15 (per block)
  const int halfsel = (lane >= 16) ? 1 : 0;

  // ---- load + hi/lo-split the 64x256 query tile into LDS (once per WG) ----
  {
    const int row = t >> 2, seg = t & 3;
    const float4* p4 = reinterpret_cast<const float4*>(pred);
    #pragma unroll
    for (int i = 0; i < 16; ++i) {
      float4 v = p4[(size_t)(wg * MTILE + row) * 64 + seg * 16 + i];
      int o = row * DDIM + seg * 64 + i * 4;
      float f[4] = {v.x, v.y, v.z, v.w};
      #pragma unroll
      for (int j = 0; j < 4; ++j) {
        unsigned short h = f32_to_bf16_rn(f[j]);
        Xhi[o + j] = h;
        Xlo[o + j] = f32_to_bf16_rn(f[j] - bf16_bits_to_f32(h));
      }
    }
  }
  __syncthreads();

  // ---- hoist A fragments into VGPRs: 8 K-steps x (hi,lo) = 128 VGPRs ----
  const int am   = wr * 16 + (lane & 15);   // A-layout: M = lane&15
  const int aoff = halfsel ? 16 : 0;        // A half-lane K offset
  bf16x16 Ah[8], Al[8];
  #pragma unroll
  for (int s = 0; s < 8; ++s) {
    Frag fh, fl;
    fh.q[0] = *reinterpret_cast<const uint4*>(&Xhi[am * DDIM + s * 32 + aoff]);
    fh.q[1] = *reinterpret_cast<const uint4*>(&Xhi[am * DDIM + s * 32 + 8 + aoff]);
    fl.q[0] = *reinterpret_cast<const uint4*>(&Xlo[am * DDIM + s * 32 + aoff]);
    fl.q[1] = *reinterpret_cast<const uint4*>(&Xlo[am * DDIM + s * 32 + 8 + aoff]);
    Ah[s] = fh.v; Al[s] = fl.v;
  }

  float best_s[8];
  int   best_k[8];
  #pragma unroll
  for (int r = 0; r < 8; ++r) { best_s[r] = 3.4e38f; best_k[r] = 0; }

  const int bn   = wc * 16 + (lane & 15);   // B-layout: N = lane&15
  const int boff = halfsel ? 16 : 0;        // B half-lane K offset

#if USE_ASYNC_LDS
  // ===================== async (ASYNCcnt) staging pipeline ==================
  // One async b128 per lane moves 16 bytes; 4 calls cover one 16 KB half.
  auto issue_block = [&](int cb, int buf) {
    gvint4* gh = (gvint4*)(eh + (size_t)cb * NBLK * DDIM);
    gvint4* gl = (gvint4*)(el + (size_t)cb * NBLK * DDIM);
    lvint4* lh = (lvint4*)(&Ehi[buf][0]);
    lvint4* ll = (lvint4*)(&Elo[buf][0]);
    #pragma unroll
    for (int i = 0; i < 4; ++i) {
      __builtin_amdgcn_global_load_async_to_lds_b128(gh + t + i * 256,
                                                     lh + t + i * 256, 0, 0);
      __builtin_amdgcn_global_load_async_to_lds_b128(gl + t + i * 256,
                                                     ll + t + i * 256, 0, 0);
    }
  };

  issue_block(0, 0);
  wait_async_lds();
  __syncthreads();

  for (int cb = 0; cb < NITER; ++cb) {
    const int cur = cb & 1;
    if (cb + 1 < NITER) issue_block(cb + 1, cur ^ 1);   // overlap with compute

    f32x8 acc = {0.f, 0.f, 0.f, 0.f, 0.f, 0.f, 0.f, 0.f};
    #pragma unroll
    for (int s = 0; s < 8; ++s) {
      const int d0 = s * 32;
      Frag bh, bl;
      bh.q[0] = *reinterpret_cast<const uint4*>(&Ehi[cur][bn * DDIM + d0 + boff]);
      bh.q[1] = *reinterpret_cast<const uint4*>(&Ehi[cur][bn * DDIM + d0 + boff + 8]);
      bl.q[0] = *reinterpret_cast<const uint4*>(&Elo[cur][bn * DDIM + d0 + boff]);
      bl.q[1] = *reinterpret_cast<const uint4*>(&Elo[cur][bn * DDIM + d0 + boff + 8]);
      acc = __builtin_amdgcn_wmma_f32_16x16x32_bf16(false, Ah[s], false, bh.v,
                                                    (short)0, acc, false, false);
      acc = __builtin_amdgcn_wmma_f32_16x16x32_bf16(false, Ah[s], false, bl.v,
                                                    (short)0, acc, false, false);
      acc = __builtin_amdgcn_wmma_f32_16x16x32_bf16(false, Al[s], false, bh.v,
                                                    (short)0, acc, false, false);
    }

    const int   kglob = cb * NBLK + wc * 16 + (lane & 15);
    const float e2v   = e2g[kglob];
    #pragma unroll
    for (int r = 0; r < 8; ++r) {
      float sc = e2v - 2.0f * acc[r];
      if (sc < best_s[r]) { best_s[r] = sc; best_k[r] = kglob; }
    }

    wait_async_lds();     // my async writes to buf[cur^1] landed
    __syncthreads();      // everyone's staging + reads of buf[cur] complete
  }
#else
  // ============ register-pipelined staging (batched b128 loads) =============
  uint4 rh[4], rl[4];
  {
    const uint4* sh = reinterpret_cast<const uint4*>(eh);
    const uint4* sl = reinterpret_cast<const uint4*>(el);
    #pragma unroll
    for (int i = 0; i < 4; ++i) {   // 8 independent loads in flight
      rh[i] = sh[t + i * 256];
      rl[i] = sl[t + i * 256];
    }
  }

  for (int cb = 0; cb < NITER; ++cb) {
    const int cur = cb & 1;
    {
      uint4* dh = reinterpret_cast<uint4*>(&Ehi[cur][0]);
      uint4* dl = reinterpret_cast<uint4*>(&Elo[cur][0]);
      #pragma unroll
      for (int i = 0; i < 4; ++i) {
        dh[t + i * 256] = rh[i];
        dl[t + i * 256] = rl[i];
      }
    }
    if (cb + 1 < NITER) {
      const uint4* sh = reinterpret_cast<const uint4*>(eh + (size_t)(cb + 1) * NBLK * DDIM);
      const uint4* sl = reinterpret_cast<const uint4*>(el + (size_t)(cb + 1) * NBLK * DDIM);
      #pragma unroll
      for (int i = 0; i < 4; ++i) {
        rh[i] = sh[t + i * 256];
        rl[i] = sl[t + i * 256];
      }
    }
    __syncthreads();   // buf[cur] visible to all waves

    f32x8 acc = {0.f, 0.f, 0.f, 0.f, 0.f, 0.f, 0.f, 0.f};
    #pragma unroll
    for (int s = 0; s < 8; ++s) {
      const int d0 = s * 32;
      Frag bh, bl;
      bh.q[0] = *reinterpret_cast<const uint4*>(&Ehi[cur][bn * DDIM + d0 + boff]);
      bh.q[1] = *reinterpret_cast<const uint4*>(&Ehi[cur][bn * DDIM + d0 + boff + 8]);
      bl.q[0] = *reinterpret_cast<const uint4*>(&Elo[cur][bn * DDIM + d0 + boff]);
      bl.q[1] = *reinterpret_cast<const uint4*>(&Elo[cur][bn * DDIM + d0 + boff + 8]);
      acc = __builtin_amdgcn_wmma_f32_16x16x32_bf16(false, Ah[s], false, bh.v,
                                                    (short)0, acc, false, false);
      acc = __builtin_amdgcn_wmma_f32_16x16x32_bf16(false, Ah[s], false, bl.v,
                                                    (short)0, acc, false, false);
      acc = __builtin_amdgcn_wmma_f32_16x16x32_bf16(false, Al[s], false, bh.v,
                                                    (short)0, acc, false, false);
    }

    const int   kglob = cb * NBLK + wc * 16 + (lane & 15);
    const float e2v   = e2g[kglob];
    #pragma unroll
    for (int r = 0; r < 8; ++r) {
      float sc = e2v - 2.0f * acc[r];
      if (sc < best_s[r]) { best_s[r] = sc; best_k[r] = kglob; }
    }
  }
#endif

  // ---- argmin reduce across the 16 lanes sharing each row ----
  #pragma unroll
  for (int r = 0; r < 8; ++r) {
    #pragma unroll
    for (int m = 8; m >= 1; m >>= 1) {
      float os = __shfl_xor(best_s[r], m, 32);
      int   ok = __shfl_xor(best_k[r], m, 32);
      if (os < best_s[r] || (os == best_s[r] && ok < best_k[r])) {
        best_s[r] = os; best_k[r] = ok;
      }
    }
  }
  if ((lane & 15) == 0) {
    const int rowbase = wr * 16 + (halfsel ? 8 : 0);
    #pragma unroll
    for (int r = 0; r < 8; ++r) {
      red_s[(rowbase + r) * 2 + wc] = best_s[r];
      red_i[(rowbase + r) * 2 + wc] = best_k[r];
    }
  }
  __syncthreads();
  if (t < MTILE) {
    float s0 = red_s[t * 2],  s1 = red_s[t * 2 + 1];
    int   k0 = red_i[t * 2],  k1 = red_i[t * 2 + 1];
    int cls = (s1 < s0 || (s1 == s0 && k1 < k0)) ? k1 : k0;
    cls_lds[t] = cls;
    out_classes[wg * MTILE + t] = (float)cls;
  }
  __syncthreads();

  // ---- gather: out[row] = embeds[cls[row]] ----
  const float4* e4 = reinterpret_cast<const float4*>(embeds);
  float4* o4 = reinterpret_cast<float4*>(out_embeds);
  #pragma unroll
  for (int i = 0; i < 16; ++i) {
    int flat = i * 256 + t;
    int row = flat >> 6, col = flat & 63;
    o4[(size_t)(wg * MTILE + row) * 64 + col] =
        e4[(size_t)cls_lds[row] * 64 + col];
  }
}

// ---------------------------------------------------------------------------
// Fallback fused kernel (no workspace): in-loop E conversion + LDS e2 atomics.
// ---------------------------------------------------------------------------
__global__ __launch_bounds__(256) void vq_argmin_fused(
    const float* __restrict__ pred, const float* __restrict__ embeds,
    float* __restrict__ out_embeds, float* __restrict__ out_classes)
{
  __shared__ __align__(16) unsigned short Xhi[MTILE * DDIM];
  __shared__ __align__(16) unsigned short Xlo[MTILE * DDIM];
  __shared__ __align__(16) unsigned short Ehi[NBLK * DDIM];
  __shared__ __align__(16) unsigned short Elo[NBLK * DDIM];
  __shared__ float e2_lds[NBLK];
  __shared__ float red_s[MTILE * 2];
  __shared__ int   red_i[MTILE * 2];
  __shared__ int   cls_lds[MTILE];

  const int t       = threadIdx.x;
  const int wg      = blockIdx.x;
  const int lane    = t & 31;
  const int wave    = t >> 5;
  const int wr      = wave >> 1;
  const int wc      = wave & 1;
  const int halfsel = (lane >= 16) ? 1 : 0;

  {
    const int row = t >> 2, seg = t & 3;
    const float4* p4 = reinterpret_cast<const float4*>(pred);
    #pragma unroll
    for (int i = 0; i < 16; ++i) {
      float4 v = p4[(size_t)(wg * MTILE + row) * 64 + seg * 16 + i];
      int o = row * DDIM + seg * 64 + i * 4;
      float f[4] = {v.x, v.y, v.z, v.w};
      #pragma unroll
      for (int j = 0; j < 4; ++j) {
        unsigned short h = f32_to_bf16_rn(f[j]);
        Xhi[o + j] = h;
        Xlo[o + j] = f32_to_bf16_rn(f[j] - bf16_bits_to_f32(h));
      }
    }
  }
  __syncthreads();

  const int am   = wr * 16 + (lane & 15);
  const int aoff = halfsel ? 16 : 0;
  bf16x16 Ah[8], Al[8];
  #pragma unroll
  for (int s = 0; s < 8; ++s) {
    Frag fh, fl;
    fh.q[0] = *reinterpret_cast<const uint4*>(&Xhi[am * DDIM + s * 32 + aoff]);
    fh.q[1] = *reinterpret_cast<const uint4*>(&Xhi[am * DDIM + s * 32 + 8 + aoff]);
    fl.q[0] = *reinterpret_cast<const uint4*>(&Xlo[am * DDIM + s * 32 + aoff]);
    fl.q[1] = *reinterpret_cast<const uint4*>(&Xlo[am * DDIM + s * 32 + 8 + aoff]);
    Ah[s] = fh.v; Al[s] = fl.v;
  }

  float best_s[8];
  int   best_k[8];
  #pragma unroll
  for (int r = 0; r < 8; ++r) { best_s[r] = 3.4e38f; best_k[r] = 0; }

  const float4* e4 = reinterpret_cast<const float4*>(embeds);
  const int bn   = wc * 16 + (lane & 15);
  const int boff = halfsel ? 16 : 0;

  for (int cb = 0; cb < NITER; ++cb) {
    __syncthreads();
    if (t < NBLK) e2_lds[t] = 0.0f;
    __syncthreads();

    if (cb + 1 < NITER) {
      const char* np = reinterpret_cast<const char*>(
          embeds + (size_t)((cb + 1) * NBLK + (t >> 3)) * DDIM + (t & 7) * 32);
      __builtin_prefetch(np, 0, 1);      // global_prefetch_b8
    }

    {
      const int c = t >> 3, chunk = t & 7;
      const int k = cb * NBLK + c;
      float ss = 0.0f;
      #pragma unroll
      for (int i = 0; i < 8; ++i) {
        float4 v = e4[(size_t)k * 64 + chunk * 8 + i];
        int o = c * DDIM + chunk * 32 + i * 4;
        float f[4] = {v.x, v.y, v.z, v.w};
        #pragma unroll
        for (int j = 0; j < 4; ++j) {
          unsigned short h = f32_to_bf16_rn(f[j]);
          Ehi[o + j] = h;
          Elo[o + j] = f32_to_bf16_rn(f[j] - bf16_bits_to_f32(h));
          ss += f[j] * f[j];
        }
      }
      atomicAdd(&e2_lds[c], ss);         // ds_add_f32
    }
    __syncthreads();

    f32x8 acc = {0.f, 0.f, 0.f, 0.f, 0.f, 0.f, 0.f, 0.f};
    #pragma unroll
    for (int s = 0; s < 8; ++s) {
      const int d0 = s * 32;
      Frag bh, bl;
      bh.q[0] = *reinterpret_cast<const uint4*>(&Ehi[bn * DDIM + d0 + boff]);
      bh.q[1] = *reinterpret_cast<const uint4*>(&Ehi[bn * DDIM + d0 + boff + 8]);
      bl.q[0] = *reinterpret_cast<const uint4*>(&Elo[bn * DDIM + d0 + boff]);
      bl.q[1] = *reinterpret_cast<const uint4*>(&Elo[bn * DDIM + d0 + boff + 8]);
      acc = __builtin_amdgcn_wmma_f32_16x16x32_bf16(false, Ah[s], false, bh.v,
                                                    (short)0, acc, false, false);
      acc = __builtin_amdgcn_wmma_f32_16x16x32_bf16(false, Ah[s], false, bl.v,
                                                    (short)0, acc, false, false);
      acc = __builtin_amdgcn_wmma_f32_16x16x32_bf16(false, Al[s], false, bh.v,
                                                    (short)0, acc, false, false);
    }

    const int   kglob = cb * NBLK + wc * 16 + (lane & 15);
    const float e2v   = e2_lds[wc * 16 + (lane & 15)];
    #pragma unroll
    for (int r = 0; r < 8; ++r) {
      float sc = e2v - 2.0f * acc[r];
      if (sc < best_s[r]) { best_s[r] = sc; best_k[r] = kglob; }
    }
  }

  #pragma unroll
  for (int r = 0; r < 8; ++r) {
    #pragma unroll
    for (int m = 8; m >= 1; m >>= 1) {
      float os = __shfl_xor(best_s[r], m, 32);
      int   ok = __shfl_xor(best_k[r], m, 32);
      if (os < best_s[r] || (os == best_s[r] && ok < best_k[r])) {
        best_s[r] = os; best_k[r] = ok;
      }
    }
  }
  if ((lane & 15) == 0) {
    const int rowbase = wr * 16 + (halfsel ? 8 : 0);
    #pragma unroll
    for (int r = 0; r < 8; ++r) {
      red_s[(rowbase + r) * 2 + wc] = best_s[r];
      red_i[(rowbase + r) * 2 + wc] = best_k[r];
    }
  }
  __syncthreads();
  if (t < MTILE) {
    float s0 = red_s[t * 2],  s1 = red_s[t * 2 + 1];
    int   k0 = red_i[t * 2],  k1 = red_i[t * 2 + 1];
    int cls = (s1 < s0 || (s1 == s0 && k1 < k0)) ? k1 : k0;
    cls_lds[t] = cls;
    out_classes[wg * MTILE + t] = (float)cls;
  }
  __syncthreads();

  float4* o4 = reinterpret_cast<float4*>(out_embeds);
  #pragma unroll
  for (int i = 0; i < 16; ++i) {
    int flat = i * 256 + t;
    int row = flat >> 6, col = flat & 63;
    o4[(size_t)(wg * MTILE + row) * 64 + col] =
        e4[(size_t)cls_lds[row] * 64 + col];
  }
}

extern "C" void kernel_launch(void* const* d_in, const int* in_sizes, int n_in,
                              void* d_out, int out_size, void* d_ws, size_t ws_size,
                              hipStream_t stream) {
  const float* pred   = (const float*)d_in[0];   // [64,32,32,256] f32
  const float* embeds = (const float*)d_in[1];   // [1024,256] f32
  float* out     = (float*)d_out;                // [NQ*DDIM] gathered embeds
  float* out_cls = out + (size_t)NQ * DDIM;      // [NQ] classes (as float)
  (void)in_sizes; (void)n_in; (void)out_size;

  const size_t ehi_bytes = (size_t)KCODE * DDIM * sizeof(unsigned short);
  const size_t need = 2 * ehi_bytes + (size_t)KCODE * sizeof(float);

  if (ws_size >= need && d_ws != nullptr) {
    unsigned short* eh = (unsigned short*)d_ws;
    unsigned short* el = (unsigned short*)((char*)d_ws + ehi_bytes);
    float*          e2 = (float*)((char*)d_ws + 2 * ehi_bytes);
    vq_prep_embeds<<<KCODE, 256, 0, stream>>>(embeds, eh, el, e2);
    vq_argmin_fast<<<NQ / MTILE, 256, 0, stream>>>(pred, embeds, eh, el, e2,
                                                   out, out_cls);
  } else {
    vq_argmin_fused<<<NQ / MTILE, 256, 0, stream>>>(pred, embeds, out, out_cls);
  }
}
